// Seq2Seq_30167850287132
// MI455X (gfx1250) — matrix operand order repaired and checked by reading, not compiled, not run on hardware
//
#include <hip/hip_runtime.h>
#include <hip/hip_bf16.h>

// ---------------------------------------------------------------------------
// Seq2Seq (bi-LSTM encoder + Bahdanau attention decoder) for gfx1250.
// All GEMMs via v_wmma_f32_16x16x32_f16 (f16 operands, f32 accum).
// B=512, S=64, T=120, E=192, H=256, EH=512, OUT_V=512.
// ---------------------------------------------------------------------------

typedef _Float16 h16;
typedef __attribute__((ext_vector_type(8)))  _Float16 v8h_t;
typedef __attribute__((ext_vector_type(16))) _Float16 v16h;
typedef __attribute__((ext_vector_type(8)))  float    v8f;

#define SB 512   // batch
#define SS 64    // src len
#define ST 120   // trg len
#define SE 192   // emb dim
#define SH 256   // hidden
#define SEH 512  // 2*H
#define SV 512   // out vocab

// ---- WMMA fragment helpers (CDNA5 16x16x32 f16 layouts, ISA 7.12.2) -------
// A 16x32: lanes 0-15 row m=lane, halves 0..7 = K[0..7], 8..15 = K[16..23];
//          lanes 16-31 row m=lane-16, K[8..15] and K[24..31].  (koff trick)
__device__ __forceinline__ v16h a_frag(const h16* p) {
  v8h_t a0 = *(const v8h_t*)(p);
  v8h_t a1 = *(const v8h_t*)(p + 16);
  v16h a;
#pragma unroll
  for (int i = 0; i < 8; ++i) { a[i] = a0[i]; a[i + 8] = a1[i]; }
  return a;
}
// B 32x16 pre-packed: lane = 16*h + n_local, 16 halves = k_local in a_frag's k order.
__device__ __forceinline__ v16h b_frag(const h16* Bp, int nt, int kt, int nKT, int lane) {
  return *(const v16h*)(Bp + ((((size_t)nt * nKT + kt) * 32 + lane) << 4));
}
__device__ __forceinline__ float sigf(float x) { return 1.f / (1.f + __expf(-x)); }

// ---- weight packer: gather up to 3 column regions of row-major f32 W ------
// B element (k,n) = Wsrc[n][col(k)], stored at ((nt*nKT+kt)*32 + lane)*16 + s, lane = 16*((k%16)/8) + n%16, s = (k%8) + 8*((k%32)/16)
__global__ void s2s_pack_b3(const float* __restrict__ W0, int ld0, int c0, int s0,
                            const float* __restrict__ W1, int ld1, int c1, int s1,
                            const float* __restrict__ W2, int ld2, int c2,
                            h16* __restrict__ out, int N, int K) {
  int i = blockIdx.x * 256 + threadIdx.x;
  if (i >= N * K) return;
  int j  = (i >> 4) & 15;
  int kl = (i & 7) + (((i >> 3) & 1) << 4) + (((i >> 8) & 1) << 3);
  int nKT = K >> 5;
  int kt = (i >> 9) % nKT;
  int nt = (i >> 9) / nKT;
  int n = nt * 16 + j;
  int k = kt * 32 + kl;
  float v;
  if (k < s0)            v = W0[(size_t)n * ld0 + c0 + k];
  else if (k < s0 + s1)  v = W1[(size_t)n * ld1 + c1 + (k - s0)];
  else                   v = W2[(size_t)n * ld2 + c2 + (k - s0 - s1)];
  out[i] = (h16)v;
}

__global__ void s2s_zero(float* __restrict__ p, int n) {
  int i = blockIdx.x * 256 + threadIdx.x;
  if (i < n) p[i] = 0.f;
}

// ---- generic WMMA GEMM: C = act(A@B + bias), optional f16 copy -----------
__global__ void s2s_gemm(const h16* __restrict__ A, int lda,
                         const h16* __restrict__ Bp,
                         float* __restrict__ C, int ldc,
                         h16* __restrict__ C16, int ldc16,
                         const float* __restrict__ bias,
                         int act, int M, int N, int K) {
  int lane = threadIdx.x & 31;
  int wid = blockIdx.x * 8 + (threadIdx.x >> 5);
  int NT = N >> 4;
  if (wid >= (M >> 4) * NT) return;
  int mt = wid / NT, nt = wid % NT;
  int nKT = K >> 5;
  int m = mt * 16 + (lane & 15);
  int koff = (lane >> 4) << 3;
  const h16* arow = A + (size_t)m * lda + koff;
  v8f acc = {};
  for (int kt = 0; kt < nKT; ++kt) {
    if (kt + 1 < nKT) __builtin_prefetch((const void*)(arow + (kt + 1) * 32), 0, 1);
    v16h a = a_frag(arow + kt * 32);
    v16h b = b_frag(Bp, nt, kt, nKT, lane);
    acc = __builtin_amdgcn_wmma_f32_16x16x32_f16(false, a, false, b, (short)0, acc, false, false);
  }
  int nl = lane & 15, mb = (lane >> 4) << 3;
  int gn = nt * 16 + nl;
  float bv = bias ? bias[gn] : 0.f;
#pragma unroll
  for (int r = 0; r < 8; ++r) {
    int gm = mt * 16 + mb + r;
    float v = acc[r] + bv;
    if (act) v = tanhf(v);
    C[(size_t)gm * ldc + gn] = v;
    if (C16) C16[(size_t)gm * ldc16 + gn] = (h16)v;
  }
}

// ---- encoder: gather this step's embeddings into encA[parity] ------------
// encA layout: [b][ dir*448 + (x:192 | h:256) ], ld = 896 halves
__global__ void s2s_enc_copyx(const int* __restrict__ src, const float* __restrict__ emb,
                              h16* __restrict__ encA, int t) {
  int i = blockIdx.x * 256 + threadIdx.x;
  if (i >= 2 * SB * SE) return;
  int j = i % SE;
  int b = (i / SE) % SB;
  int d = i / (SE * SB);
  int tidx = d ? (SS - 1 - t) : t;
  int tok = src[b * SS + tidx];
  encA[(size_t)b * 896 + d * 448 + j] = (h16)emb[(size_t)tok * SE + j];
}

// ---- fused encoder LSTM step: gates GEMM (K=448) + cell update, both dirs -
__global__ void s2s_enc_step(const h16* __restrict__ cur, h16* __restrict__ nxt,
                             const h16* __restrict__ packF, const h16* __restrict__ packB,
                             const float* __restrict__ biasF, const float* __restrict__ biasB,
                             float* __restrict__ c_state, h16* __restrict__ enc_outs16, int t) {
  int lane = threadIdx.x & 31;
  int wid = blockIdx.x * 8 + (threadIdx.x >> 5);   // 1024 waves: dir(2) x mt(32) x ntg(16)
  int d = wid >> 9, mt = (wid >> 4) & 31, ntg = wid & 15;
  const h16* Bp = d ? packB : packF;
  const float* bias = d ? biasB : biasF;
  const int nKT = 14;                               // K = 448
  int m = mt * 16 + (lane & 15);
  int koff = (lane >> 4) << 3;
  const h16* arow = cur + (size_t)m * 896 + d * 448 + koff;
  v8f acc[4] = {};
  for (int kt = 0; kt < nKT; ++kt) {
    v16h a = a_frag(arow + kt * 32);
#pragma unroll
    for (int g = 0; g < 4; ++g) {
      v16h b = b_frag(Bp, g * 16 + ntg, kt, nKT, lane);
      acc[g] = __builtin_amdgcn_wmma_f32_16x16x32_f16(false, a, false, b, (short)0, acc[g], false, false);
    }
  }
  int nl = lane & 15, mb = (lane >> 4) << 3;
  int n = ntg * 16 + nl;
  int tidx = d ? (SS - 1 - t) : t;
#pragma unroll
  for (int r = 0; r < 8; ++r) {
    int b = mt * 16 + mb + r;
    float gi = acc[0][r] + bias[n];
    float gf = acc[1][r] + bias[256 + n];
    float gg = acc[2][r] + bias[512 + n];
    float go = acc[3][r] + bias[768 + n];
    size_t ci = ((size_t)d * SB + b) * SH + n;
    float c = sigf(gf) * c_state[ci] + sigf(gi) * tanhf(gg);
    c_state[ci] = c;
    float h = sigf(go) * tanhf(c);
    enc_outs16[((size_t)b * SS + tidx) * SEH + d * SH + n] = (h16)h;
    nxt[(size_t)b * 896 + d * 448 + SE + n] = (h16)h;
  }
}

// ---- concat final fwd/bwd hidden for enc_fc -------------------------------
__global__ void s2s_hcat(const h16* __restrict__ encA0, h16* __restrict__ hcat) {
  int i = blockIdx.x * 256 + threadIdx.x;
  if (i >= SB * 512) return;
  int j = i & 511, b = i >> 9;
  int col = (j < SH) ? (SE + j) : (448 + SE + (j - SH));
  hcat[i] = encA0[(size_t)b * 896 + col];
}

// ---- attention + context + decoder-input staging (one block per batch b) --
// xcat layout: [b][ e:192 | ctx:512 | h:256 ], ld = 960 halves
__global__ void s2s_attn(const float* __restrict__ enc_proj, const float* __restrict__ attn_h,
                         const h16* __restrict__ enc_outs16, const float* __restrict__ attn_v,
                         const int* __restrict__ src, const int* __restrict__ trg,
                         const float* __restrict__ dec_emb,
                         h16* __restrict__ xcur, h16* __restrict__ xnxt, int t) {
  int b = blockIdx.x, tid = threadIdx.x;
  int lane = tid & 31, w = tid >> 5;
  __shared__ float sh_h[SH];
  __shared__ float sh_sc[SS];
  __shared__ float sh_aw[SS];
  for (int j = tid; j < SH; j += 256) sh_h[j] = attn_h[b * SH + j];
  __syncthreads();
  for (int s = w; s < SS; s += 8) {
    const float* ep = enc_proj + ((size_t)b * SS + s) * SH;
    float acc = 0.f;
    for (int j = lane; j < SH; j += 32) acc += tanhf(ep[j] + sh_h[j]) * attn_v[j];
#pragma unroll
    for (int off = 16; off; off >>= 1) acc += __shfl_xor(acc, off, 32);
    if (lane == 0) sh_sc[s] = (src[b * SS + s] != 0) ? acc : -1e9f;
  }
  __syncthreads();
  if (w == 0) {
    float m = fmaxf(sh_sc[lane], sh_sc[lane + 32]);
#pragma unroll
    for (int off = 16; off; off >>= 1) m = fmaxf(m, __shfl_xor(m, off, 32));
    float e0 = __expf(sh_sc[lane] - m), e1 = __expf(sh_sc[lane + 32] - m);
    float ss = e0 + e1;
#pragma unroll
    for (int off = 16; off; off >>= 1) ss += __shfl_xor(ss, off, 32);
    float inv = 1.f / ss;
    sh_aw[lane] = e0 * inv;
    sh_aw[lane + 32] = e1 * inv;
  }
  __syncthreads();
  const h16* eo = enc_outs16 + (size_t)b * SS * SEH;
  for (int k = tid; k < SEH; k += 256) {
    float acc = 0.f;
#pragma unroll 4
    for (int s = 0; s < SS; ++s) acc += sh_aw[s] * (float)eo[s * SEH + k];
    h16 hv = (h16)acc;
    xcur[(size_t)b * 960 + SE + k] = hv;
    xnxt[(size_t)b * 960 + SE + k] = hv;
  }
  int tok = (t == 0) ? 1 : trg[b * ST + (t - 1)];
  for (int j = tid; j < SE; j += 256) {
    h16 ev = (h16)dec_emb[(size_t)tok * SE + j];
    xcur[(size_t)b * 960 + j] = ev;
    xnxt[(size_t)b * 960 + j] = ev;
  }
}

// ---- fused decoder LSTM step: gates GEMM (K=960 = e+ctx+h) + cell update --
__global__ void s2s_dec_step(const h16* __restrict__ cur, h16* __restrict__ nxt,
                             const h16* __restrict__ Bp, const float* __restrict__ bias,
                             float* __restrict__ c_state) {
  int lane = threadIdx.x & 31;
  int wid = blockIdx.x * 8 + (threadIdx.x >> 5);   // 512 waves: mt(32) x ntg(16)
  int mt = wid >> 4, ntg = wid & 15;
  const int nKT = 30;                               // K = 960
  int m = mt * 16 + (lane & 15);
  int koff = (lane >> 4) << 3;
  const h16* arow = cur + (size_t)m * 960 + koff;
  v8f acc[4] = {};
  for (int kt = 0; kt < nKT; ++kt) {
    v16h a = a_frag(arow + kt * 32);
#pragma unroll
    for (int g = 0; g < 4; ++g) {
      v16h b = b_frag(Bp, g * 16 + ntg, kt, nKT, lane);
      acc[g] = __builtin_amdgcn_wmma_f32_16x16x32_f16(false, a, false, b, (short)0, acc[g], false, false);
    }
  }
  int nl = lane & 15, mb = (lane >> 4) << 3;
  int n = ntg * 16 + nl;
#pragma unroll
  for (int r = 0; r < 8; ++r) {
    int b = mt * 16 + mb + r;
    float gi = acc[0][r] + bias[n];
    float gf = acc[1][r] + bias[256 + n];
    float gg = acc[2][r] + bias[512 + n];
    float go = acc[3][r] + bias[768 + n];
    size_t ci = (size_t)b * SH + n;
    float c = sigf(gf) * c_state[ci] + sigf(gi) * tanhf(gg);
    c_state[ci] = c;
    float h = sigf(go) * tanhf(c);
    nxt[(size_t)b * 960 + 704 + n] = (h16)h;
  }
}

// ---------------------------------------------------------------------------
extern "C" void kernel_launch(void* const* d_in, const int* in_sizes, int n_in,
                              void* d_out, int out_size, void* d_ws, size_t ws_size,
                              hipStream_t stream) {
  (void)in_sizes; (void)n_in; (void)out_size; (void)ws_size;
  const int*   src     = (const int*)d_in[0];
  const int*   trg     = (const int*)d_in[2];
  const float* enc_emb = (const float*)d_in[3];
  const float* Wih_f   = (const float*)d_in[4];
  const float* Whh_f   = (const float*)d_in[5];
  const float* b_f     = (const float*)d_in[6];
  const float* Wih_b   = (const float*)d_in[7];
  const float* Whh_b   = (const float*)d_in[8];
  const float* b_b     = (const float*)d_in[9];
  const float* fc_W    = (const float*)d_in[10];
  const float* fc_b    = (const float*)d_in[11];
  const float* dec_emb = (const float*)d_in[12];
  const float* dec_Wih = (const float*)d_in[13];
  const float* dec_Whh = (const float*)d_in[14];
  const float* dec_b   = (const float*)d_in[15];
  const float* attn_W  = (const float*)d_in[16];
  const float* attn_b  = (const float*)d_in[17];
  const float* attn_v  = (const float*)d_in[18];
  const float* out_W   = (const float*)d_in[19];
  const float* out_b   = (const float*)d_in[20];
  float* out = (float*)d_out;

  char* ws = (char*)d_ws;
  size_t off = 0;
  auto alloc = [&](size_t bytes) -> char* {
    char* p = ws + off;
    off += (bytes + 255) & ~(size_t)255;
    return p;
  };
  h16* pGatesF   = (h16*)alloc((size_t)1024 * 448 * 2);
  h16* pGatesB   = (h16*)alloc((size_t)1024 * 448 * 2);
  h16* pFc       = (h16*)alloc((size_t)256 * 512 * 2);
  h16* pDec      = (h16*)alloc((size_t)1024 * 960 * 2);
  h16* pAttnH    = (h16*)alloc((size_t)256 * 256 * 2);
  h16* pAttnE    = (h16*)alloc((size_t)256 * 512 * 2);
  h16* pOut      = (h16*)alloc((size_t)512 * 960 * 2);
  h16* enc_outs16= (h16*)alloc((size_t)SB * SS * SEH * 2);   // 33.5 MB
  float* enc_proj= (float*)alloc((size_t)SB * SS * SH * 4);  // 33.5 MB
  h16* encA0     = (h16*)alloc((size_t)SB * 896 * 2);
  h16* encA1     = (h16*)alloc((size_t)SB * 896 * 2);
  h16* hcat      = (h16*)alloc((size_t)SB * 512 * 2);
  float* c_enc   = (float*)alloc((size_t)2 * SB * SH * 4);
  h16* xcat0     = (h16*)alloc((size_t)SB * 960 * 2);
  h16* xcat1     = (h16*)alloc((size_t)SB * 960 * 2);
  float* dec_c   = (float*)alloc((size_t)SB * SH * 4);
  float* attn_hb = (float*)alloc((size_t)SB * SH * 4);

  // ---- state init (h0 = c0 = 0) ----
  s2s_zero<<<896, 256, 0, stream>>>((float*)encA0, SB * 896 / 2);
  s2s_zero<<<1024, 256, 0, stream>>>(c_enc, 2 * SB * SH);
  s2s_zero<<<512, 256, 0, stream>>>(dec_c, SB * SH);

  // ---- pack weights into WMMA B-fragment layouts ----
  s2s_pack_b3<<<1792, 256, 0, stream>>>(Wih_f, 192, 0, 192, Whh_f, 256, 0, 256,
                                        Wih_f, 192, 0, pGatesF, 1024, 448);
  s2s_pack_b3<<<1792, 256, 0, stream>>>(Wih_b, 192, 0, 192, Whh_b, 256, 0, 256,
                                        Wih_b, 192, 0, pGatesB, 1024, 448);
  s2s_pack_b3<<<512, 256, 0, stream>>>(fc_W, 512, 0, 512, fc_W, 512, 0, 0,
                                       fc_W, 512, 0, pFc, 256, 512);
  s2s_pack_b3<<<3840, 256, 0, stream>>>(dec_Wih, 704, 0, 704, dec_Whh, 256, 0, 256,
                                        dec_Wih, 704, 0, pDec, 1024, 960);
  s2s_pack_b3<<<256, 256, 0, stream>>>(attn_W, 768, 0, 256, attn_W, 768, 0, 0,
                                       attn_W, 768, 0, pAttnH, 256, 256);
  s2s_pack_b3<<<512, 256, 0, stream>>>(attn_W, 768, 256, 512, attn_W, 768, 0, 0,
                                       attn_W, 768, 0, pAttnE, 256, 512);
  // pred A = [e | ctx | h] -> permute out_W columns: e<-768.., ctx<-256.., h<-0..
  s2s_pack_b3<<<1920, 256, 0, stream>>>(out_W, 960, 768, 192, out_W, 960, 256, 512,
                                        out_W, 960, 0, pOut, 512, 960);

  // ---- encoder recurrence (both directions fused per step) ----
  for (int t = 0; t < SS; ++t) {
    h16* cur = (t & 1) ? encA1 : encA0;
    h16* nxt = (t & 1) ? encA0 : encA1;
    s2s_enc_copyx<<<768, 256, 0, stream>>>(src, enc_emb, cur, t);
    s2s_enc_step<<<128, 256, 0, stream>>>(cur, nxt, pGatesF, pGatesB, b_f, b_b,
                                          c_enc, enc_outs16, t);
  }
  s2s_hcat<<<1024, 256, 0, stream>>>(encA0, hcat);
  // h0 = tanh(hcat @ fc_W.T + fc_b) -> f16 into xcat0 h-region (f32 scratch = attn_hb)
  s2s_gemm<<<64, 256, 0, stream>>>(hcat, 512, pFc, attn_hb, SH, xcat0 + 704, 960,
                                   fc_b, 1, SB, SH, 512);
  // enc_proj = enc_outs @ attn_W[:,H:].T + attn_b   (step-invariant part)
  s2s_gemm<<<4096, 256, 0, stream>>>(enc_outs16, SEH, pAttnE, enc_proj, SH,
                                     (h16*)nullptr, 0, attn_b, 0, SB * SS, SH, 512);

  // ---- decoder ----
  for (int t = 0; t < ST; ++t) {
    h16* cur = (t & 1) ? xcat1 : xcat0;
    h16* nxt = (t & 1) ? xcat0 : xcat1;
    // attn_h = h @ attn_W[:,:H].T
    s2s_gemm<<<64, 256, 0, stream>>>(cur + 704, 960, pAttnH, attn_hb, SH,
                                     (h16*)nullptr, 0, (const float*)nullptr, 0,
                                     SB, SH, SH);
    s2s_attn<<<SB, 256, 0, stream>>>(enc_proj, attn_hb, enc_outs16, attn_v,
                                     src, trg, dec_emb, cur, nxt, t);
    s2s_dec_step<<<64, 256, 0, stream>>>(cur, nxt, pDec, dec_b, dec_c);
    // pred = [h,ctx,e] @ out_W.T + out_b  -> d_out[:, t, :]
    s2s_gemm<<<128, 256, 0, stream>>>(nxt, 960, pOut, out + (size_t)t * SV, ST * SV,
                                      (h16*)nullptr, 0, out_b, 0, SB, SV, 960);
  }
}
